// NONA_NN_79336635892439
// MI455X (gfx1250) — matrix operand order repaired and checked by reading, not compiled, not run on hardware
//
#include <hip/hip_runtime.h>
#include <math.h>

// Problem constants (match reference)
#define B_Q   2048
#define N_N   32768
#define D_F   512
#define K_TOP 64
#define C_CLS 100

// Tiling
#define TM     16              // query rows per workgroup (one WMMA M-tile)
#define CHUNK  2048            // neighbors per workgroup chunk
#define NCHUNK (N_N / CHUNK)   // 16
#define WG     256             // 8 waves (wave32)

typedef __attribute__((ext_vector_type(16))) __bf16 v16bf;
typedef __attribute__((ext_vector_type(8)))  __bf16 v8bf;
typedef __attribute__((ext_vector_type(8)))  float  v8f;

static __device__ __forceinline__ unsigned short f2bf(float f) {
  // round-to-nearest-even float -> bf16
  unsigned u = __float_as_uint(f);
  u += 0x7FFFu + ((u >> 16) & 1u);
  return (unsigned short)(u >> 16);
}

// ---------------------------------------------------------------------------
// Kernel 1: per-row squared norms (fp32, exact) + bf16 conversion.
// One wave per row of D_F=512 (16 floats per lane).
// ---------------------------------------------------------------------------
__global__ void nona_prep(const float* __restrict__ src,
                          float* __restrict__ norm,
                          unsigned short* __restrict__ dst,
                          int rows) {
  const int lane = threadIdx.x & 31;
  const int wv   = threadIdx.x >> 5;
  const int row  = blockIdx.x * (WG / 32) + wv;
  if (row >= rows) return;

  const float4* rp = (const float4*)(src + (size_t)row * D_F + lane * 16);
  float s = 0.0f;
  unsigned pk[8];
#pragma unroll
  for (int i = 0; i < 4; ++i) {
    float4 v = rp[i];
    s += v.x * v.x + v.y * v.y + v.z * v.z + v.w * v.w;
    pk[2 * i]     = (unsigned)f2bf(v.x) | ((unsigned)f2bf(v.y) << 16);
    pk[2 * i + 1] = (unsigned)f2bf(v.z) | ((unsigned)f2bf(v.w) << 16);
  }
  uint4* d4 = (uint4*)(dst + (size_t)row * D_F + lane * 16);
  d4[0] = make_uint4(pk[0], pk[1], pk[2], pk[3]);
  d4[1] = make_uint4(pk[4], pk[5], pk[6], pk[7]);

#pragma unroll
  for (int m = 16; m >= 1; m >>= 1) s += __shfl_xor(s, m, 32);
  if (lane == 0) norm[row] = s;
}

// ---------------------------------------------------------------------------
// Kernel 2: bf16 WMMA distance GEMM + exact per-chunk top-64.
// Grid: (B/TM, N/CHUNK). Block: 256 (8 waves).
//   d2[m][n] = x2[m] + xn2[n] - 2 * dot_bf16(x[m], xn[n])   (clamped >= 0)
// Each (row, chunk) emits its exact 64 smallest d2 (value + global index).
// ---------------------------------------------------------------------------
__global__ void nona_gemm_topk(const unsigned short* __restrict__ xbf,
                               const unsigned short* __restrict__ xnbf,
                               const float* __restrict__ x2,
                               const float* __restrict__ xn2,
                               float* __restrict__ cand_v,
                               int* __restrict__ cand_i) {
  extern __shared__ char smem[];
  unsigned short* As   = (unsigned short*)smem;               // TM*D_F bf16 (16 KB)
  float*          d2s  = (float*)(smem + TM * D_F * 2);       // TM*CHUNK f32 (128 KB)
  float*          x2s  = d2s + TM * CHUNK;                    // TM
  float*          xn2s = x2s + TM;                            // CHUNK (8 KB)
  unsigned*       cnts = (unsigned*)(xn2s + CHUNK);           // 16 counters

  const int tid  = threadIdx.x;
  const int lane = tid & 31;
  const int wv   = tid >> 5;
  const int m0   = blockIdx.x * TM;
  const int nb0  = blockIdx.y * CHUNK;

  // Stage A-tile (already bf16) into LDS.
  {
    const uint4* g = (const uint4*)(xbf + (size_t)m0 * D_F);
    uint4* sdst = (uint4*)As;
    for (int i = tid; i < TM * D_F / 8; i += WG) sdst[i] = g[i];
  }
  if (tid < TM) x2s[tid] = x2[m0 + tid];
  for (int i = tid; i < CHUNK; i += WG) xn2s[i] = xn2[nb0 + i];
  __syncthreads();

  // wave32 fragment layout used by this kernel (contiguous k, same for A and B):
  //  A (16x32 bf16): lane holds M=lane&15, K = (lane<16 ? 0..15 : 16..31) contiguous
  //  B (32x16 bf16): lane holds column n=lane&15, K = (lane<16 ? 0..15 : 16..31) contiguous
  //  D (16x16 f32):  element (g + 8*(lane>=16), lane&15) in acc[g]
  const int half = lane >> 4;
  const int l16  = lane & 15;
  const unsigned short* ap0 = As + l16 * D_F + half * 16;

  for (int t = wv; t < CHUNK / 16; t += WG / 32) {
    const int nloc = t * 16 + l16;
    const unsigned short* bp = xnbf + (size_t)(nb0 + nloc) * D_F + half * 16;
    __builtin_prefetch(bp, 0, 0);     // global_prefetch_b8 on the B stream
    v8f acc = {};
#pragma unroll
    for (int kc = 0; kc < D_F / 32; ++kc) {
      v8bf alo = *(const v8bf*)(ap0 + kc * 32);
      v8bf ahi = *(const v8bf*)(ap0 + kc * 32 + 8);
      v16bf a = __builtin_shufflevector(alo, ahi,
                 0, 1, 2, 3, 4, 5, 6, 7, 8, 9, 10, 11, 12, 13, 14, 15);
      v16bf b = *(const v16bf*)(bp + kc * 32);
      acc = __builtin_amdgcn_wmma_f32_16x16x32_bf16(false, a, false, b,
                                                    (short)0, acc, false, false);
    }
    const float xn2v = xn2s[nloc];
#pragma unroll
    for (int g = 0; g < 8; ++g) {
      const int m = g + half * 8;
      const float d2v = x2s[m] + xn2v - 2.0f * acc[g];
      d2s[m * CHUNK + nloc] = fmaxf(d2v, 0.0f);   // clamp: uint order == float order
    }
  }
  __syncthreads();

  // Exact top-64 smallest d2 per row within this chunk.
  // Wave wv handles rows wv and wv+8; bit-pattern binary search for the
  // 64th-smallest value T, then collect (<T) plus quota-limited (==T).
  for (int rr = 0; rr < 2; ++rr) {
    const int r = wv + rr * (WG / 32);
    const float* rowv = d2s + r * CHUNK;

    unsigned lo = 0u, hi = 0x7F800000u;
    while (lo < hi) {
      const unsigned mid = lo + ((hi - lo) >> 1);
      int c = 0;
      for (int j = lane; j < CHUNK; j += 32)
        c += (__float_as_uint(rowv[j]) <= mid) ? 1 : 0;
#pragma unroll
      for (int s = 16; s >= 1; s >>= 1) c += __shfl_xor(c, s, 32);
      if (c >= K_TOP) hi = mid; else lo = mid + 1;
    }
    const unsigned T = lo;

    unsigned* c_lt = &cnts[wv * 2];
    unsigned* c_eq = &cnts[wv * 2 + 1];
    if (lane == 0) { atomicExch(c_lt, 0u); atomicExch(c_eq, 0u); }
    // LDS ops from one wave complete in order (wave32 single PC) -> no barrier needed.
    const size_t base = ((size_t)(m0 + r) * NCHUNK + blockIdx.y) * K_TOP;
    for (int j = lane; j < CHUNK; j += 32) {
      const float v = rowv[j];
      if (__float_as_uint(v) < T) {            // count(<T) <= 63 by construction
        const unsigned slot = atomicAdd(c_lt, 1u);
        cand_v[base + slot] = v;
        cand_i[base + slot] = nb0 + j;
      }
    }
    const unsigned c1 = atomicAdd(c_lt, 0u);
    for (int j = lane; j < CHUNK; j += 32) {
      const float v = rowv[j];
      if (__float_as_uint(v) == T) {
        const unsigned slot = c1 + atomicAdd(c_eq, 1u);
        if (slot < K_TOP) { cand_v[base + slot] = v; cand_i[base + slot] = nb0 + j; }
      }
    }
  }
}

// ---------------------------------------------------------------------------
// Kernel 3: per row, merge 16*64 candidates -> exact global top-64,
// softmax over the 64 survivors, scatter-add one-hot labels, clip.
// One workgroup per query row.
// ---------------------------------------------------------------------------
__global__ void nona_merge(const float* __restrict__ cand_v,
                           const int* __restrict__ cand_i,
                           const int* __restrict__ y_n,
                           float* __restrict__ out) {
  __shared__ float vals[NCHUNK * K_TOP];   // 1024
  __shared__ int   idxs[NCHUNK * K_TOP];
  __shared__ float pval[K_TOP];
  __shared__ int   pidx[K_TOP];
  __shared__ float outacc[C_CLS];
  __shared__ unsigned c_lt, c_eq;

  const int tid  = threadIdx.x;
  const int lane = tid & 31;
  const int wv   = tid >> 5;
  const int row  = blockIdx.x;
  const int M    = NCHUNK * K_TOP;

  for (int i = tid; i < M; i += WG) {
    vals[i] = cand_v[(size_t)row * M + i];
    idxs[i] = cand_i[(size_t)row * M + i];
  }
  for (int i = tid; i < C_CLS; i += WG) outacc[i] = 0.0f;
  __syncthreads();

  if (wv == 0) {
    unsigned lo = 0u, hi = 0x7F800000u;
    while (lo < hi) {
      const unsigned mid = lo + ((hi - lo) >> 1);
      int c = 0;
      for (int j = lane; j < M; j += 32)
        c += (__float_as_uint(vals[j]) <= mid) ? 1 : 0;
#pragma unroll
      for (int s = 16; s >= 1; s >>= 1) c += __shfl_xor(c, s, 32);
      if (c >= K_TOP) hi = mid; else lo = mid + 1;
    }
    const unsigned T = lo;

    if (lane == 0) { atomicExch(&c_lt, 0u); atomicExch(&c_eq, 0u); }
    for (int j = lane; j < M; j += 32) {
      const float v = vals[j];
      if (__float_as_uint(v) < T) {
        const unsigned s = atomicAdd(&c_lt, 1u);
        pval[s] = v; pidx[s] = idxs[j];
      }
    }
    const unsigned c1 = atomicAdd(&c_lt, 0u);
    for (int j = lane; j < M; j += 32) {
      const float v = vals[j];
      if (__float_as_uint(v) == T) {
        const unsigned s = c1 + atomicAdd(&c_eq, 1u);
        if (s < K_TOP) { pval[s] = v; pidx[s] = idxs[j]; }
      }
    }

    // softmax over the 64 kept similarities: sim = -sqrt(max(d2,0)+1e-12)
    const float s0 = -sqrtf(fmaxf(pval[lane], 0.0f) + 1e-12f);
    const float s1 = -sqrtf(fmaxf(pval[lane + 32], 0.0f) + 1e-12f);
    float mx = fmaxf(s0, s1);
#pragma unroll
    for (int s = 16; s >= 1; s >>= 1) mx = fmaxf(mx, __shfl_xor(mx, s, 32));
    const float e0 = __expf(s0 - mx);
    const float e1 = __expf(s1 - mx);
    float sum = e0 + e1;
#pragma unroll
    for (int s = 16; s >= 1; s >>= 1) sum += __shfl_xor(sum, s, 32);
    const float inv = 1.0f / sum;
    atomicAdd(&outacc[y_n[pidx[lane]]],      e0 * inv);
    atomicAdd(&outacc[y_n[pidx[lane + 32]]], e1 * inv);
  }
  __syncthreads();

  for (int c = tid; c < C_CLS; c += WG)
    out[(size_t)row * C_CLS + c] = fminf(fmaxf(outacc[c], 0.0f), 1.0f);
}

// ---------------------------------------------------------------------------
// Host launcher. Workspace layout (~50.2 MB total):
//   x2[B] f32 | xn2[N] f32 | x bf16 [B*D] | xn bf16 [N*D] |
//   cand_v [B*16*64] f32 | cand_i [B*16*64] i32
// ---------------------------------------------------------------------------
extern "C" void kernel_launch(void* const* d_in, const int* in_sizes, int n_in,
                              void* d_out, int out_size, void* d_ws, size_t ws_size,
                              hipStream_t stream) {
  (void)in_sizes; (void)n_in; (void)out_size; (void)ws_size;
  const float* x  = (const float*)d_in[0];
  const float* xn = (const float*)d_in[1];
  const int*   yn = (const int*)d_in[2];
  float* out = (float*)d_out;

  char* ws = (char*)d_ws;
  size_t off = 0;
  float* x2 = (float*)(ws + off);                      off += (size_t)B_Q * 4;
  float* xn2 = (float*)(ws + off);                     off += (size_t)N_N * 4;
  unsigned short* xbf  = (unsigned short*)(ws + off);  off += (size_t)B_Q * D_F * 2;
  unsigned short* xnbf = (unsigned short*)(ws + off);  off += (size_t)N_N * D_F * 2;
  float* cand_v = (float*)(ws + off);                  off += (size_t)B_Q * NCHUNK * K_TOP * 4;
  int*   cand_i = (int*)(ws + off);                    off += (size_t)B_Q * NCHUNK * K_TOP * 4;

  nona_prep<<<B_Q / (WG / 32), WG, 0, stream>>>(x,  x2,  xbf,  B_Q);
  nona_prep<<<N_N / (WG / 32), WG, 0, stream>>>(xn, xn2, xnbf, N_N);

  const size_t smem2 = (size_t)TM * D_F * 2      // A tile bf16
                     + (size_t)TM * CHUNK * 4    // d2 stage f32
                     + (size_t)TM * 4            // x2 tile
                     + (size_t)CHUNK * 4         // xn2 chunk
                     + 16 * 4;                   // counters  => 155,776 B (<320KB/WGP)
  hipFuncSetAttribute(reinterpret_cast<const void*>(nona_gemm_topk),
                      hipFuncAttributeMaxDynamicSharedMemorySize, (int)smem2);
  nona_gemm_topk<<<dim3(B_Q / TM, NCHUNK), WG, smem2, stream>>>(
      xbf, xnbf, x2, xn2, cand_v, cand_i);

  nona_merge<<<B_Q, WG, 0, stream>>>(cand_v, cand_i, yn, out);
}
